// SCorrSAN_75213467287788
// MI455X (gfx1250) — matrix operand order repaired and checked by reading, not compiled, not run on hardware
//
#include <hip/hip_runtime.h>
#include <hip/hip_bf16.h>

// ---------------------------------------------------------------------------
// SCorrSAN pipeline on MI455X (gfx1250).
//   * bf16 v_wmma_f32_16x16x32_bf16 for the SCE 1x1-conv GEMM and the
//     correlation GEMM.
//   * TDM (tensor_load_to_lds + s_wait_tensorcnt) stages the per-block shared
//     A panel into LDS; A fragments then come from ds_load_b128.
//   * fused resize+softmax+soft-argmax kernel avoids materializing the 268 MB
//     fine-grid correlation tensor.
// Workspace (offsets 256-aligned): Wbf 2048x1056 bf16 | Xbf 8x1024x1056 bf16 |
//   nrm 8x1024 f32 | Ybf 8x1024x2048 bf16 | corr 4x1024x1024 f32 | smax/tmax.
// ---------------------------------------------------------------------------

typedef __attribute__((ext_vector_type(8)))  float          v8f;
typedef __attribute__((ext_vector_type(16))) __bf16         v16bf;
typedef __attribute__((ext_vector_type(8)))  unsigned short v8us;
typedef __attribute__((ext_vector_type(16))) unsigned short v16us;
typedef __attribute__((ext_vector_type(4)))  unsigned int   v4u;
typedef __attribute__((ext_vector_type(8)))  int            v8i;
typedef __attribute__((ext_vector_type(4)))  int            v4i;

union V16U { v16us u; v16bf b; };

__device__ __forceinline__ unsigned short f2bf(float f) {
  unsigned u = __float_as_uint(f);
  u += 0x7FFFu + ((u >> 16) & 1u);           // round-to-nearest-even
  return (unsigned short)(u >> 16);
}
__device__ __forceinline__ float bf2f(unsigned short h) {
  return __uint_as_float(((unsigned)h) << 16);
}

// A-fragment (16x32 bf16) from a row-major panel (LDS or global):
// lane L holds row M=L%16, K = (L/16)*16 .. +15 contiguous (matches B).
__device__ __forceinline__ v16bf load_frag_A(const unsigned short* base,
                                             int rowStride, int row0, int k0) {
  int lane = threadIdx.x & 31;
  const unsigned short* p =
      base + (long)(row0 + (lane & 15)) * rowStride + k0 + ((lane >> 4) << 4);
  V16U t;
  t.u = *(const v16us*)(p);
  return t.b;
}

// B-fragment (32x16 bf16) with B[k][n] = src[n][k]: lane L holds column
// N=L%16, K = (L/16)*16 .. +15 contiguous (this code's order, same as A).
__device__ __forceinline__ v16bf load_frag_B(const unsigned short* base,
                                             int rowStride, int row0, int k0) {
  int lane = threadIdx.x & 31;
  V16U t;
  t.u = *(const v16us*)(base + (long)(row0 + (lane & 15)) * rowStride + k0 +
                        ((lane >> 4) << 4));
  return t.b;
}

// --- TDM: stage a contiguous 2-D bf16 tile (tileRows x rowElems, full-width,
// row stride rowElems) from global into LDS. One issue per workgroup (wave 0),
// then s_wait_tensorcnt + barrier. D# layout per CDNA5 ISA ch.8.
// Toolchain here exposes the 6-arg builtin:
//   (uint32x4 g0, int32x8 g1, int32x4 g2, int32x4 g3, int32x8 pad, i32 cpol)
__device__ __forceinline__ void tdm_stage_panel(unsigned short* smem,
                                                const unsigned short* gsrc,
                                                unsigned rowElems,
                                                unsigned tileRows) {
#if __has_builtin(__builtin_amdgcn_tensor_load_to_lds)
  if ((threadIdx.x >> 5) == 0) {
    unsigned long ga = (unsigned long)(uintptr_t)gsrc;        // byte address
    unsigned lo = (unsigned)(uintptr_t)smem;                  // LDS byte offset
    v4u g0;
    g0[0] = 1u;                                               // count=1, normal
    g0[1] = lo;                                               // lds_addr
    g0[2] = (unsigned)(ga & 0xFFFFFFFFu);                     // global_addr lo
    g0[3] = (unsigned)((ga >> 32) & 0x1FFFFFFu) | (2u << 30); // addr hi | type=2
    v8i g1;
    g1[0] = (int)(1u << 16);                  // data_size=1 (2 bytes/elem)
    g1[1] = (int)((rowElems & 0xFFFFu) << 16);          // tensor_dim0[15:0]
    g1[2] = (int)((rowElems >> 16) | (tileRows << 16)); // td0 hi | td1 lo
    g1[3] = (int)((rowElems & 0xFFFFu) << 16);          // td1 hi(0) | tile_dim0
    g1[4] = (int)tileRows;                    // tile_dim1 | tile_dim2=0
    g1[5] = (int)rowElems;                    // tensor_dim0_stride[31:0]
    g1[6] = 0;                                // stride0 hi | stride1 lo
    g1[7] = 0;
    v4i z4 = {0, 0, 0, 0};
    v8i z8 = {0, 0, 0, 0, 0, 0, 0, 0};
    __builtin_amdgcn_tensor_load_to_lds(g0, g1, z4, z4, z8, 0);
    __builtin_amdgcn_s_wait_tensorcnt(0);
  }
  __syncthreads();
#else
  unsigned nvec = rowElems * tileRows / 8;    // contiguous full-width panel
  for (unsigned i = threadIdx.x; i < nvec; i += blockDim.x)
    ((v8us*)smem)[i] = ((const v8us*)gsrc)[i];
  __syncthreads();
#endif
}

// --- weights f32 (2048x1052) -> bf16 (2048x1056, zero pad) -----------------
__global__ void k_wbf(const float* __restrict__ w, unsigned short* __restrict__ wbf) {
  int i = blockIdx.x * 256 + threadIdx.x;          // 2048*1056
  int row = i / 1056, col = i - row * 1056;
  float v = (col < 1052) ? w[(long)row * 1052 + col] : 0.f;
  wbf[i] = f2bf(v);
}

// --- per-position feature L2 norm (clamped at 1e-12) -----------------------
__global__ void k_nrm(const float* __restrict__ src, const float* __restrict__ tgt,
                      float* __restrict__ nrm) {
  int t = blockIdx.x * 256 + threadIdx.x;          // 8*1024
  int g = t >> 10, pos = t & 1023;
  const float* f = (g < 4) ? src + (long)g * 1048576 : tgt + (long)(g - 4) * 1048576;
  float s = 0.f;
  for (int c = 0; c < 1024; ++c) { float v = f[(long)c * 1024 + pos]; s += v * v; }
  nrm[t] = fmaxf(sqrtf(s), 1e-12f);
}

// --- transpose feat (ch,pos) -> Xbf (pos,ch) in bf16 via LDS tiles ---------
__global__ void k_xpose(const float* __restrict__ src, const float* __restrict__ tgt,
                        unsigned short* __restrict__ xbf) {
  __shared__ float tile[32][33];
  int bid = blockIdx.x;                            // 8*32*32
  int g = bid >> 10, ct = (bid >> 5) & 31, pt = bid & 31;
  const float* f = (g < 4) ? src + (long)g * 1048576 : tgt + (long)(g - 4) * 1048576;
  int tx = threadIdx.x & 31, ty = threadIdx.x >> 5;
  for (int r = 0; r < 32; r += 8)
    tile[ty + r][tx] = f[(long)(ct * 32 + ty + r) * 1024 + pt * 32 + tx];
  __syncthreads();
  unsigned short* xg = xbf + (long)g * 1024 * 1056;
  for (int r = 0; r < 32; r += 8)
    xg[(long)(pt * 32 + ty + r) * 1056 + ct * 32 + tx] = f2bf(tile[tx][ty + r]);
}

// --- 28 shifted local correlations (+4 zero-pad channels) ------------------
__global__ void k_ctx(const float* __restrict__ src, const float* __restrict__ tgt,
                      const float* __restrict__ nrm, unsigned short* __restrict__ xbf) {
  long i = (long)blockIdx.x * 256 + threadIdx.x;   // 8*32*1024
  int pos = (int)(i & 1023);
  int o   = (int)((i >> 10) & 31);
  int g   = (int)(i >> 15);
  unsigned short* out = xbf + (long)g * 1024 * 1056 + (long)pos * 1056 + 1024 + o;
  if (o >= 28) { *out = 0; return; }
  int r, c;
  if (o < 7)       { r = o;      c = o; }
  else if (o < 14) { r = o - 7;  c = 3; }
  else if (o < 21) { r = o - 14; c = 6 - (o - 14); }
  else             { r = 3;      c = o - 21; }
  int y = pos >> 5, x = pos & 31;
  int y2 = y + r - 3, x2 = x + c - 3;
  float res = 0.f;
  if (y2 >= 0 && y2 < 32 && x2 >= 0 && x2 < 32) {
    const float* f = (g < 4) ? src + (long)g * 1048576 : tgt + (long)(g - 4) * 1048576;
    int pos2 = y2 * 32 + x2;
    float s = 0.f;
    for (int ch = 0; ch < 1024; ++ch)
      s += f[(long)ch * 1024 + pos] * f[(long)ch * 1024 + pos2];
    res = s / (nrm[g * 1024 + pos] * nrm[g * 1024 + pos2]);
  }
  *out = f2bf(res);
}

// --- SCE 1x1 conv GEMM: Y[pos, chOut] = relu(X[pos,:1056] W[chOut,:]^T + b)
// block = 4 waves sharing one 16x1056 A panel (TDM-staged in LDS);
// each wave computes a 16(pos) x 64(chOut) strip, K=1056 = 33 wmma steps.
__global__ void __launch_bounds__(128)
k_gemm1(const unsigned short* __restrict__ xbf, const unsigned short* __restrict__ wbf,
        const float* __restrict__ bias, unsigned short* __restrict__ ybf) {
  __shared__ unsigned short sA[16 * 1056];         // 33 KB
  int bid = blockIdx.x;                            // 8g * 64mt * 8
  int g = bid >> 9;
  int m0 = ((bid >> 3) & 63) << 4;
  int wave = threadIdx.x >> 5;
  int n0 = (((bid & 7) << 2) + wave) << 6;
  const unsigned short* xg = xbf + (long)g * 1024 * 1056;
  tdm_stage_panel(sA, xg + (long)m0 * 1056, 1056u, 16u);
  v8f acc[4] = {};
  for (int kt = 0; kt < 33; ++kt) {
    int k0 = kt * 32;
    if (kt < 32)
      __builtin_prefetch(wbf + (long)n0 * 1056 + k0 + 32, 0, 3);
    v16bf a = load_frag_A(sA, 1056, 0, k0);        // ds_load path
#pragma unroll
    for (int j = 0; j < 4; ++j) {
      v16bf b = load_frag_B(wbf, 1056, n0 + j * 16, k0);
      acc[j] = __builtin_amdgcn_wmma_f32_16x16x32_bf16(
          false, a, false, b, (short)0, acc[j], false, false);
    }
  }
  int lane = threadIdx.x & 31;
  int nlo = lane & 15, madd = (lane >> 4) << 3;
  unsigned short* yg = ybf + (long)g * 1024 * 2048;
#pragma unroll
  for (int j = 0; j < 4; ++j) {
    int n = n0 + j * 16 + nlo;
    float bv = bias[n];
#pragma unroll
    for (int r = 0; r < 8; ++r) {
      float y = fmaxf(acc[j][r] + bv, 0.f);
      yg[(long)(m0 + madd + r) * 2048 + n] = f2bf(y);
    }
  }
}

// --- FeatureL2Norm over 2048 channels, in place on bf16 --------------------
__global__ void k_l2norm(unsigned short* __restrict__ ybf) {
  int row = blockIdx.x * 8 + (threadIdx.x >> 5);   // 8192 rows
  int lane = threadIdx.x & 31;
  unsigned short* p = ybf + (long)row * 2048;
  float s = 0.f;
  for (int c = lane; c < 2048; c += 32) { float v = bf2f(p[c]); s += v * v; }
  for (int off = 16; off; off >>= 1) s += __shfl_xor(s, off, 32);
  float sc = rsqrtf(s + 1e-6f);
  for (int c = lane; c < 2048; c += 32) p[c] = f2bf(bf2f(p[c]) * sc);
}

// --- correlation GEMM: corr[b][s][t] = sum_ch S[s,ch]*T[t,ch], K = 2048 ----
// block = 4 waves sharing one 16x2048 S panel (TDM-staged in LDS).
__global__ void __launch_bounds__(128)
k_gemm2(const unsigned short* __restrict__ ybf, float* __restrict__ corr) {
  __shared__ unsigned short sA[16 * 2048];         // 64 KB
  int bid = blockIdx.x;                            // 4b * 64mt * 4
  int b = bid >> 8;
  int m0 = ((bid >> 2) & 63) << 4;
  int wave = threadIdx.x >> 5;
  int n0 = (((bid & 3) << 2) + wave) << 6;
  const unsigned short* S = ybf + (long)b * 1024 * 2048;
  const unsigned short* T = ybf + (long)(4 + b) * 1024 * 2048;
  tdm_stage_panel(sA, S + (long)m0 * 2048, 2048u, 16u);
  v8f acc[4] = {};
  for (int kt = 0; kt < 64; ++kt) {
    int k0 = kt * 32;
    if (kt < 63)
      __builtin_prefetch(T + (long)n0 * 2048 + k0 + 32, 0, 3);
    v16bf a = load_frag_A(sA, 2048, 0, k0);        // ds_load path
#pragma unroll
    for (int j = 0; j < 4; ++j) {
      v16bf bb = load_frag_B(T, 2048, n0 + j * 16, k0);
      acc[j] = __builtin_amdgcn_wmma_f32_16x16x32_bf16(
          false, a, false, bb, (short)0, acc[j], false, false);
    }
  }
  int lane = threadIdx.x & 31;
  int nlo = lane & 15, madd = (lane >> 4) << 3;
  float* cb = corr + (long)b * 1024 * 1024;
#pragma unroll
  for (int j = 0; j < 4; ++j)
#pragma unroll
    for (int r = 0; r < 8; ++r)
      cb[(long)(m0 + madd + r) * 1024 + n0 + j * 16 + nlo] = acc[j][r];
}

// --- mutual NN filter: row max, col max, then corr^3/(smax*tmax) -----------
__global__ void k_rowmax(const float* __restrict__ corr, float* __restrict__ smax) {
  int row = blockIdx.x * 8 + (threadIdx.x >> 5);   // 4096
  int lane = threadIdx.x & 31;
  const float* p = corr + (long)row * 1024;
  float m = -3.4e38f;
  for (int t = lane; t < 1024; t += 32) m = fmaxf(m, p[t]);
  for (int off = 16; off; off >>= 1) m = fmaxf(m, __shfl_xor(m, off, 32));
  if (lane == 0) smax[row] = m;
}

__global__ void k_colmax(const float* __restrict__ corr, float* __restrict__ tmax) {
  int id = blockIdx.x * 256 + threadIdx.x;         // 4096
  int b = id >> 10, t = id & 1023;
  const float* p = corr + (long)b * 1024 * 1024 + t;
  float m = -3.4e38f;
  for (int s = 0; s < 1024; ++s) m = fmaxf(m, p[(long)s * 1024]);
  tmax[id] = m;
}

__global__ void k_mutual(float* __restrict__ corr, const float* __restrict__ smax,
                         const float* __restrict__ tmax) {
  long i = (long)blockIdx.x * 256 + threadIdx.x;   // 4*1024*1024
  int t = (int)(i & 1023);
  int s = (int)((i >> 10) & 1023);
  int b = (int)(i >> 20);
  float v = corr[i];
  float sm = smax[b * 1024 + s]; sm = (sm == 0.f) ? 1e-30f : sm;
  float tm = tmax[b * 1024 + t]; tm = (tm == 0.f) ? 1e-30f : tm;
  corr[i] = v * (v / sm) * (v / tm);
}

// --- fused: bilinear resize (src & tgt grids) + softmax(beta) + soft-argmax
__global__ void __launch_bounds__(256)
k_flow(const float* __restrict__ corr, float* __restrict__ out) {
  __shared__ float mbuf[1024];
  __shared__ float cbuf[4096];
  __shared__ float red[256];
  int tid = threadIdx.x;
  int bb = blockIdx.x;                             // 4*4096
  int b = bb >> 12, tt = bb & 4095;
  int yt = tt >> 6, xt = tt & 63;
  const float scale = 31.f / 63.f;                 // align_corners=True, 32->64
  float fy = yt * scale; int iy0 = (int)fy; float wy = fy - iy0; int iy1 = min(iy0 + 1, 31);
  float fx = xt * scale; int ix0 = (int)fx; float wx = fx - ix0; int ix1 = min(ix0 + 1, 31);
  float wA = (1.f - wy) * (1.f - wx), wB = (1.f - wy) * wx;
  float wC = wy * (1.f - wx),         wD = wy * wx;
  int tA = iy0 * 32 + ix0, tB = iy0 * 32 + ix1, tC = iy1 * 32 + ix0, tD = iy1 * 32 + ix1;
  const float* cb = corr + (long)b * 1024 * 1024;
  for (int s = tid; s < 1024; s += 256) {
    const float* row = cb + (long)s * 1024;
    mbuf[s] = wA * row[tA] + wB * row[tB] + wC * row[tC] + wD * row[tD];
  }
  __syncthreads();
  float lmax = -3.4e38f;
  for (int ss = tid; ss < 4096; ss += 256) {
    int ys = ss >> 6, xs = ss & 63;
    float gy = ys * scale; int jy0 = (int)gy; float vy = gy - jy0; int jy1 = min(jy0 + 1, 31);
    float gx = xs * scale; int jx0 = (int)gx; float vx = gx - jx0; int jx1 = min(jx0 + 1, 31);
    float c = (1.f - vy) * (1.f - vx) * mbuf[jy0 * 32 + jx0]
            + (1.f - vy) * vx         * mbuf[jy0 * 32 + jx1]
            + vy * (1.f - vx)         * mbuf[jy1 * 32 + jx0]
            + vy * vx                 * mbuf[jy1 * 32 + jx1];
    cbuf[ss] = c;
    lmax = fmaxf(lmax, c);
  }
  red[tid] = lmax; __syncthreads();
  for (int s2 = 128; s2; s2 >>= 1) {
    if (tid < s2) red[tid] = fmaxf(red[tid], red[tid + s2]);
    __syncthreads();
  }
  float cmax = red[0]; __syncthreads();
  float zacc = 0.f, xacc = 0.f, yacc = 0.f;
  for (int ss = tid; ss < 4096; ss += 256) {
    int ys = ss >> 6, xs = ss & 63;
    float e = __expf((cbuf[ss] - cmax) * 50.f);    // 1/beta = 50
    zacc += e;
    xacc += e * (-1.f + xs * (2.f / 63.f));
    yacc += e * (-1.f + ys * (2.f / 63.f));
  }
  red[tid] = zacc; __syncthreads();
  for (int s2 = 128; s2; s2 >>= 1) { if (tid < s2) red[tid] += red[tid + s2]; __syncthreads(); }
  float Z = red[0]; __syncthreads();
  red[tid] = xacc; __syncthreads();
  for (int s2 = 128; s2; s2 >>= 1) { if (tid < s2) red[tid] += red[tid + s2]; __syncthreads(); }
  float GX = red[0]; __syncthreads();
  red[tid] = yacc; __syncthreads();
  for (int s2 = 128; s2; s2 >>= 1) { if (tid < s2) red[tid] += red[tid + s2]; __syncthreads(); }
  float GY = red[0];
  if (tid == 0) {
    float gx = GX / Z, gy = GY / Z;
    float mx = (gx + 1.f) * 31.5f;                 // (W-1)/2
    float my = (gy + 1.f) * 31.5f;
    out[((long)b * 2 + 0) * 4096 + tt] = mx - (float)xt;
    out[((long)b * 2 + 1) * 4096 + tt] = my - (float)yt;
  }
}

// ---------------------------------------------------------------------------
extern "C" void kernel_launch(void* const* d_in, const int* in_sizes, int n_in,
                              void* d_out, int out_size, void* d_ws, size_t ws_size,
                              hipStream_t stream) {
  const float* src  = (const float*)d_in[0];
  const float* tgt  = (const float*)d_in[1];
  const float* sceW = (const float*)d_in[2];
  const float* sceB = (const float*)d_in[3];
  float* out = (float*)d_out;

  char* ws = (char*)d_ws;
  unsigned short* Wbf = (unsigned short*)(ws);                 //  4,325,376 B
  unsigned short* Xbf = (unsigned short*)(ws + 4325376);       // 17,301,504 B
  float*          nrm = (float*)(ws + 21626880);               //     32,768 B
  unsigned short* Ybf = (unsigned short*)(ws + 21659648);      // 33,554,432 B
  float*          cor = (float*)(ws + 55214080);               // 16,777,216 B
  float*          sM  = (float*)(ws + 71991296);               //     16,384 B
  float*          tM  = (float*)(ws + 72007680);               //     16,384 B

  k_wbf   <<<8448,  256, 0, stream>>>(sceW, Wbf);
  k_nrm   <<<32,    256, 0, stream>>>(src, tgt, nrm);
  k_xpose <<<8192,  256, 0, stream>>>(src, tgt, Xbf);
  k_ctx   <<<1024,  256, 0, stream>>>(src, tgt, nrm, Xbf);
  k_gemm1 <<<4096,  128, 0, stream>>>(Xbf, Wbf, sceB, Ybf);
  k_l2norm<<<1024,  256, 0, stream>>>(Ybf);
  k_gemm2 <<<1024,  128, 0, stream>>>(Ybf, cor);
  k_rowmax<<<512,   256, 0, stream>>>(cor, sM);
  k_colmax<<<16,    256, 0, stream>>>(cor, tM);
  k_mutual<<<16384, 256, 0, stream>>>(cor, sM, tM);
  k_flow  <<<16384, 256, 0, stream>>>(cor, out);
}
